// GATNodeClassification_7421703487980
// MI455X (gfx1250) — matrix operand order repaired and checked by reading, not compiled, not run on hardware
//
#include <hip/hip_runtime.h>

// ---------------------------------------------------------------------------
// GAT 2-layer forward for MI455X (gfx1250, wave32).
//  - GEMMs via v_wmma_f32_16x16x32_bf16 (bf16 A/B, fp32 accum) - memory bound,
//    so bf16 streaming + free matrix pipes is the right operating point.
//  - Edge phase (dominant cost: ~1.7M edges x 512B gather + 512B scatter-add)
//    via wave-per-edge float4 gathers and L2 float atomics.
//  - GEMM epilogue: single base address per col-tile, immediate-offset b32
//    stores, unguarded fast path for full 16-row tiles (N % 16 == 0).
// ---------------------------------------------------------------------------

#define NEG_SLOPE 0.2f

typedef __attribute__((ext_vector_type(16))) __bf16 v16bf;
typedef __attribute__((ext_vector_type(8)))  __bf16 v8bf;
typedef __attribute__((ext_vector_type(4)))  __bf16 v4bf;
typedef __attribute__((ext_vector_type(8)))  float  v8f;

// ---- WMMA fragment loaders (16-bit, 16x32 A / 32x16 B, wave32 layouts) ----
// A (16x32, row-major rows in memory): lanes 0-15 row M=lane hold K 0-7,16-23;
// lanes 16-31 row M=lane-16 hold K 8-15,24-31.
__device__ __forceinline__ v16bf load_fragA(const __bf16* __restrict__ p, int half) {
  const v8bf lo = *(const v8bf*)(p + half * 8);
  const v8bf hi = *(const v8bf*)(p + 16 + half * 8);
  v16bf r;
#pragma unroll
  for (int i = 0; i < 8; ++i) { r[i] = lo[i]; r[i + 8] = hi[i]; }
  return r;
}
// B (32x16) from pre-transposed weights [col][k]: lanes 0-15 col N=lane hold
// K 0-7,16-23; lanes 16-31 col N=lane-16 hold K 8-15,24-31.
__device__ __forceinline__ v16bf load_fragB(const __bf16* __restrict__ p, int half) {
  const v8bf lo = *(const v8bf*)(p + half * 8);
  const v8bf hi = *(const v8bf*)(p + 16 + half * 8);
  v16bf r;
#pragma unroll
  for (int i = 0; i < 8; ++i) { r[i] = lo[i]; r[i + 8] = hi[i]; }
  return r;
}

// ---- GEMM: C[nrows, LDC] = A[nrows,128] (bf16) x Bt[NCT*16,128]^T (bf16) ----
template <int NCT, int LDC>
__global__ void k_gemm_wmma(const __bf16* __restrict__ A, const __bf16* __restrict__ Bt,
                            float* __restrict__ C, int nrows) {
  const int lane = threadIdx.x & 31;
  const int wave = threadIdx.x >> 5;
  const int row0 = blockIdx.x * 128 + wave * 16;
  if (row0 >= nrows) return;
  const int m = lane & 15;
  const int half = lane >> 4;
  int arow = row0 + m;
  if (arow >= nrows) arow = nrows - 1;  // load clamp (tail tile only)

  const __bf16* ap = A + (size_t)arow * 128;
  v16bf afrag[4];
#pragma unroll
  for (int kb = 0; kb < 4; ++kb) afrag[kb] = load_fragA(ap + kb * 32, half);

  const bool full = (row0 + 16 <= nrows);
#pragma unroll
  for (int ct = 0; ct < NCT; ++ct) {
    v8f acc = {};
    const __bf16* bp = Bt + (size_t)(ct * 16 + m) * 128;
#pragma unroll
    for (int kb = 0; kb < 4; ++kb) {
      v16bf bfrag = load_fragB(bp + kb * 32, half);
      acc = __builtin_amdgcn_wmma_f32_16x16x32_bf16(
          false, afrag[kb], false, bfrag, (short)0, acc, false, false);
    }
    // C/D layout: VGPR r -> (M = r + half*8, N = m). One base pointer; all
    // row stores use immediate offsets (r*LDC*4 bytes).
    float* cp = C + (size_t)(row0 + half * 8) * LDC + ct * 16 + m;
    if (full) {
#pragma unroll
      for (int r = 0; r < 8; ++r) cp[r * LDC] = acc[r];
    } else {
#pragma unroll
      for (int r = 0; r < 8; ++r)
        if (row0 + half * 8 + r < nrows) cp[r * LDC] = acc[r];
    }
  }
}

// ---- conversions / transposes ----
// 8 elements per thread: 2x b128 loads -> 1x b128 bf16 store.
__global__ void k_cvt_bf16_v8(const float* __restrict__ in, __bf16* __restrict__ out,
                              long long n8) {
  long long i = (long long)blockIdx.x * blockDim.x + threadIdx.x;
  if (i >= n8) return;
  const float4 a = ((const float4*)in)[i * 2];
  const float4 b = ((const float4*)in)[i * 2 + 1];
  v8bf o;
  o[0] = (__bf16)a.x; o[1] = (__bf16)a.y; o[2] = (__bf16)a.z; o[3] = (__bf16)a.w;
  o[4] = (__bf16)b.x; o[5] = (__bf16)b.y; o[6] = (__bf16)b.z; o[7] = (__bf16)b.w;
  ((v8bf*)out)[i] = o;
}
__global__ void k_transpose_w1(const float* __restrict__ W, __bf16* __restrict__ Wt) {
  int i = blockIdx.x * blockDim.x + threadIdx.x;  // 128x128
  if (i >= 128 * 128) return;
  int col = i >> 7, k = i & 127;
  Wt[col * 128 + k] = (__bf16)W[k * 128 + col];
}
__global__ void k_transpose_w2(const float* __restrict__ W, __bf16* __restrict__ Wt) {
  int i = blockIdx.x * blockDim.x + threadIdx.x;  // 48x128 (pad 40->48)
  if (i >= 48 * 128) return;
  int col = i >> 7, k = i & 127;
  Wt[col * 128 + k] = (col < 40) ? (__bf16)W[k * 40 + col] : (__bf16)0.0f;
}

// ---- attention coefficients ----
__global__ void k_att_coef1(const float* __restrict__ h1, const float* __restrict__ a_src,
                            const float* __restrict__ a_dst, float* __restrict__ als,
                            float* __restrict__ ald, int n) {
  int t = blockIdx.x * blockDim.x + threadIdx.x;
  if (t >= n * 4) return;
  int node = t >> 2, hd = t & 3;
  const float* hp = h1 + (size_t)node * 128 + hd * 32;
  const float* as = a_src + hd * 32;
  const float* ad = a_dst + hd * 32;
  float s = 0.f, d = 0.f;
#pragma unroll
  for (int c = 0; c < 32; ++c) { float v = hp[c]; s += v * as[c]; d += v * ad[c]; }
  als[t] = s; ald[t] = d;
}
__global__ void k_att_coef2(const float* __restrict__ h2, const float* __restrict__ a_src,
                            const float* __restrict__ a_dst, float* __restrict__ als,
                            float* __restrict__ ald, int n) {
  int node = blockIdx.x * blockDim.x + threadIdx.x;
  if (node >= n) return;
  const float* hp = h2 + (size_t)node * 48;
  float s = 0.f, d = 0.f;
#pragma unroll
  for (int c = 0; c < 40; ++c) { float v = hp[c]; s += v * a_src[c]; d += v * a_dst[c]; }
  als[node] = s; ald[node] = d;
}

// ---- order-preserving float<->uint for atomicMax segment-max ----
__device__ __forceinline__ unsigned fflip(float f) {
  unsigned u = __float_as_uint(f);
  return (u & 0x80000000u) ? ~u : (u | 0x80000000u);
}
__device__ __forceinline__ float funflip(unsigned u) {
  return __uint_as_float((u & 0x80000000u) ? (u ^ 0x80000000u) : ~u);
}
__device__ __forceinline__ void edge_nodes(const long long* __restrict__ ei,
                                           long long e, long long E, int& src, int& dst) {
  if (e < E) { src = (int)ei[e]; dst = (int)ei[E + e]; }
  else       { src = dst = (int)(e - E); }  // virtual self-loops
}

// ---- layer-1 edge passes (4 heads) ----
__global__ void k_edge1_max(const long long* __restrict__ ei, long long E, long long Etot,
                            const float* __restrict__ als, const float* __restrict__ ald,
                            float* __restrict__ ebuf, unsigned* __restrict__ mflip) {
  long long t = (long long)blockIdx.x * blockDim.x + threadIdx.x;
  if (t >= Etot * 4) return;
  long long e = t >> 2; int hd = (int)(t & 3);
  int src, dst; edge_nodes(ei, e, E, src, dst);
  float v = als[src * 4 + hd] + ald[dst * 4 + hd];
  v = v > 0.f ? v : NEG_SLOPE * v;
  ebuf[t] = v;
  atomicMax(&mflip[dst * 4 + hd], fflip(v));
}
__global__ void k_edge1_exp(const long long* __restrict__ ei, long long E, long long Etot,
                            float* __restrict__ ebuf, const unsigned* __restrict__ mflip,
                            float* __restrict__ denom) {
  long long t = (long long)blockIdx.x * blockDim.x + threadIdx.x;
  if (t >= Etot * 4) return;
  long long e = t >> 2; int hd = (int)(t & 3);
  int src, dst; edge_nodes(ei, e, E, src, dst);
  float mx = funflip(mflip[dst * 4 + hd]);
  float ex = __expf(ebuf[t] - mx);
  ebuf[t] = ex;
  atomicAdd(&denom[dst * 4 + hd], ex);
}
__global__ void k_edge1_agg(const long long* __restrict__ ei, long long E, long long Etot,
                            const float* __restrict__ ebuf, const float* __restrict__ denom,
                            const float* __restrict__ h1, float* __restrict__ out1) {
  long long gw = ((long long)blockIdx.x * blockDim.x + threadIdx.x) >> 5;
  int lane = threadIdx.x & 31;
  if (gw >= Etot) return;
  int src, dst; edge_nodes(ei, gw, E, src, dst);
  int hd = lane >> 3;  // 8 lanes x 4ch per head
  float alpha = ebuf[gw * 4 + hd] / denom[dst * 4 + hd];
  const float4 hv = *(const float4*)(h1 + (size_t)src * 128 + lane * 4);
  float* op = out1 + (size_t)dst * 128 + lane * 4;
  atomicAdd(op + 0, alpha * hv.x);
  atomicAdd(op + 1, alpha * hv.y);
  atomicAdd(op + 2, alpha * hv.z);
  atomicAdd(op + 3, alpha * hv.w);
}

// ---- layer-1 finalize: +b1, ELU, to bf16 for GEMM2 (4 elems / thread) ----
__global__ void k_finalize1(const float* __restrict__ out1, const float* __restrict__ b1,
                            __bf16* __restrict__ hb, long long n4) {
  long long i = (long long)blockIdx.x * blockDim.x + threadIdx.x;
  if (i >= n4) return;
  const float4 v = ((const float4*)out1)[i];
  const int c = (int)((i * 4) & 127);
  float r[4] = {v.x + b1[c], v.y + b1[c + 1], v.z + b1[c + 2], v.w + b1[c + 3]};
  v4bf o;
#pragma unroll
  for (int j = 0; j < 4; ++j) {
    float e = r[j] > 0.f ? r[j] : (__expf(r[j]) - 1.f);
    o[j] = (__bf16)e;
  }
  ((v4bf*)hb)[i] = o;
}

// ---- layer-2 edge passes (1 head) ----
__global__ void k_edge2_max(const long long* __restrict__ ei, long long E, long long Etot,
                            const float* __restrict__ als, const float* __restrict__ ald,
                            float* __restrict__ ebuf, unsigned* __restrict__ mflip) {
  long long e = (long long)blockIdx.x * blockDim.x + threadIdx.x;
  if (e >= Etot) return;
  int src, dst; edge_nodes(ei, e, E, src, dst);
  float v = als[src] + ald[dst];
  v = v > 0.f ? v : NEG_SLOPE * v;
  ebuf[e] = v;
  atomicMax(&mflip[dst], fflip(v));
}
__global__ void k_edge2_exp(const long long* __restrict__ ei, long long E, long long Etot,
                            float* __restrict__ ebuf, const unsigned* __restrict__ mflip,
                            float* __restrict__ denom) {
  long long e = (long long)blockIdx.x * blockDim.x + threadIdx.x;
  if (e >= Etot) return;
  int src, dst; edge_nodes(ei, e, E, src, dst);
  float ex = __expf(ebuf[e] - funflip(mflip[dst]));
  ebuf[e] = ex;
  atomicAdd(&denom[dst], ex);
}
__global__ void k_edge2_agg(const long long* __restrict__ ei, long long E, long long Etot,
                            const float* __restrict__ ebuf, const float* __restrict__ denom,
                            const float* __restrict__ h2, float* __restrict__ out2) {
  long long gw = ((long long)blockIdx.x * blockDim.x + threadIdx.x) >> 5;
  int lane = threadIdx.x & 31;
  if (gw >= Etot) return;
  int src, dst; edge_nodes(ei, gw, E, src, dst);
  float alpha = ebuf[gw] / denom[dst];
  const float* hp = h2 + (size_t)src * 48;   // padded stride
  float* op = out2 + (size_t)dst * 40;
  atomicAdd(op + lane, alpha * hp[lane]);
  if (lane < 8) atomicAdd(op + 32 + lane, alpha * hp[32 + lane]);
}

// ---- final log_softmax over 40 classes ----
__global__ void k_logsoftmax(const float* __restrict__ out2, const float* __restrict__ b2,
                             float* __restrict__ y, int n) {
  int node = blockIdx.x * blockDim.x + threadIdx.x;
  if (node >= n) return;
  float v[40];
  float mx = -3.402823466e38f;
#pragma unroll
  for (int c = 0; c < 40; ++c) { v[c] = out2[(size_t)node * 40 + c] + b2[c]; mx = fmaxf(mx, v[c]); }
  float s = 0.f;
#pragma unroll
  for (int c = 0; c < 40; ++c) s += __expf(v[c] - mx);
  float ls = __logf(s);
#pragma unroll
  for (int c = 0; c < 40; ++c) y[(size_t)node * 40 + c] = v[c] - mx - ls;
}

static inline unsigned cdiv_u(long long a, long long b) { return (unsigned)((a + b - 1) / b); }

extern "C" void kernel_launch(void* const* d_in, const int* in_sizes, int n_in,
                              void* d_out, int out_size, void* d_ws, size_t ws_size,
                              hipStream_t stream) {
  const float*     x      = (const float*)d_in[0];
  const long long* ei     = (const long long*)d_in[1];
  const float*     W1     = (const float*)d_in[2];
  const float*     a_src1 = (const float*)d_in[3];
  const float*     a_dst1 = (const float*)d_in[4];
  const float*     b1     = (const float*)d_in[5];
  const float*     W2     = (const float*)d_in[6];
  const float*     a_src2 = (const float*)d_in[7];
  const float*     a_dst2 = (const float*)d_in[8];
  const float*     b2     = (const float*)d_in[9];
  float*           y      = (float*)d_out;

  const long long N    = in_sizes[0] / 128;
  const long long E    = in_sizes[1] / 2;
  const long long Etot = E + N;

  // ---- workspace bump allocator (256B aligned) with buffer aliasing ----
  char* ws = (char*)d_ws;
  size_t off = 0;
  auto alloc = [&](size_t bytes) -> char* {
    char* p = ws + off;
    off = (off + bytes + 255) & ~((size_t)255);
    return p;
  };
  __bf16*   xb     = (__bf16*)alloc((size_t)N * 128 * 2);   // reused as h_bf16
  __bf16*   W1t    = (__bf16*)alloc(128 * 128 * 2);
  __bf16*   W2t    = (__bf16*)alloc(48 * 128 * 2);
  float*    h1     = (float*)alloc((size_t)N * 128 * 4);    // reused as out2
  float*    out1   = (float*)alloc((size_t)N * 128 * 4);    // reused as h2
  float*    als    = (float*)alloc((size_t)N * 4 * 4);      // L1 then L2
  float*    ald    = (float*)alloc((size_t)N * 4 * 4);
  unsigned* mflip  = (unsigned*)alloc((size_t)N * 4 * 4);
  float*    denom  = (float*)alloc((size_t)N * 4 * 4);
  float*    ebuf   = (float*)alloc((size_t)Etot * 4 * 4);   // L1 (x4 heads) then L2
  __bf16*   hb     = xb;      // layer-1 output (bf16), aliases xb (dead after GEMM1)
  float*    h2     = out1;    // aliases out1 (dead after finalize1)
  float*    out2   = h1;      // aliases h1 (dead after edge1_agg)
  (void)ws_size; (void)n_in; (void)out_size;

  const int T = 256;

  // 1) precision/layout prep
  k_cvt_bf16_v8<<<cdiv_u(N * 16, T), T, 0, stream>>>(x, xb, N * 16);
  k_transpose_w1<<<cdiv_u(128 * 128, T), T, 0, stream>>>(W1, W1t);
  k_transpose_w2<<<cdiv_u(48 * 128, T), T, 0, stream>>>(W2, W2t);

  // 2) GEMM1: h1 = x @ W1  (WMMA bf16 -> f32)
  k_gemm_wmma<8, 128><<<cdiv_u(N, 128), T, 0, stream>>>(xb, W1t, h1, (int)N);

  // 3) attention logits per node/head
  k_att_coef1<<<cdiv_u(N * 4, T), T, 0, stream>>>(h1, a_src1, a_dst1, als, ald, (int)N);

  // 4) layer-1 segment softmax + aggregation
  hipMemsetAsync(mflip, 0, (size_t)N * 4 * 4, stream);     // flipped -inf
  hipMemsetAsync(denom, 0, (size_t)N * 4 * 4, stream);
  hipMemsetAsync(out1,  0, (size_t)N * 128 * 4, stream);
  k_edge1_max<<<cdiv_u(Etot * 4, T), T, 0, stream>>>(ei, E, Etot, als, ald, ebuf, mflip);
  k_edge1_exp<<<cdiv_u(Etot * 4, T), T, 0, stream>>>(ei, E, Etot, ebuf, mflip, denom);
  k_edge1_agg<<<cdiv_u(Etot * 32, T), T, 0, stream>>>(ei, E, Etot, ebuf, denom, h1, out1);

  // 5) ELU -> bf16
  k_finalize1<<<cdiv_u(N * 32, T), T, 0, stream>>>(out1, b1, hb, N * 32);

  // 6) GEMM2: h2 = h @ W2 (padded to 48 cols)
  k_gemm_wmma<3, 48><<<cdiv_u(N, 128), T, 0, stream>>>(hb, W2t, h2, (int)N);

  // 7) layer-2 attention
  k_att_coef2<<<cdiv_u(N, T), T, 0, stream>>>(h2, a_src2, a_dst2, als, ald, (int)N);
  hipMemsetAsync(mflip, 0, (size_t)N * 4, stream);
  hipMemsetAsync(denom, 0, (size_t)N * 4, stream);
  hipMemsetAsync(out2,  0, (size_t)N * 40 * 4, stream);
  k_edge2_max<<<cdiv_u(Etot, T), T, 0, stream>>>(ei, E, Etot, als, ald, ebuf, mflip);
  k_edge2_exp<<<cdiv_u(Etot, T), T, 0, stream>>>(ei, E, Etot, ebuf, mflip, denom);
  k_edge2_agg<<<cdiv_u(Etot * 32, T), T, 0, stream>>>(ei, E, Etot, ebuf, denom, h2, out2);

  // 8) log_softmax
  k_logsoftmax<<<cdiv_u(N, T), T, 0, stream>>>(out2, b2, y, (int)N);
}
